// TGAT_1271310319918
// MI455X (gfx1250) — matrix operand order repaired and checked by reading, not compiled, not run on hardware
//
#include <hip/hip_runtime.h>
#include <hip/hip_bf16.h>
#include <math.h>

#define T_STEPS 1024
#define N_NODES 33
#define N_EDGES 97
#define HID1 128
#define HID2 64

typedef __attribute__((ext_vector_type(16))) __bf16 v16bf;
typedef __attribute__((ext_vector_type(8)))  float  v8f;

__device__ __forceinline__ unsigned short f2bf(float f) {
  union { float f; unsigned u; } x; x.f = f;
  unsigned r = x.u + 0x7FFFu + ((x.u >> 16) & 1u);  // round-to-nearest-even
  return (unsigned short)(r >> 16);
}
__device__ __forceinline__ float sigm(float x) { return 1.f / (1.f + expf(-x)); }

// ---------------------------------------------------------------------------
// Prep kernels
// ---------------------------------------------------------------------------
// ep[e*512+c] = sum_d edge_attr[e*3+d] * we[d*512+c]
__global__ void edge_proj_kernel(const float* __restrict__ ea,
                                 const float* __restrict__ we,
                                 float* __restrict__ ep) {
  int e = blockIdx.x, c = threadIdx.x;
  ep[e * 512 + c] = ea[e * 3] * we[c] + ea[e * 3 + 1] * we[512 + c] + ea[e * 3 + 2] * we[1024 + c];
}

// Wt[n*512+k] = bf16(W[k*512+n])   (512x512 transpose + convert)
__global__ void convert_transpose_512(const float* __restrict__ W, unsigned short* __restrict__ Wt) {
  int n = blockIdx.x, k = threadIdx.x;
  Wt[n * 512 + k] = f2bf(W[k * 512 + n]);
}

__global__ void convert_bf16_kernel(const float* __restrict__ s, unsigned short* __restrict__ d, int n) {
  int i = blockIdx.x * blockDim.x + threadIdx.x;
  if (i < n) d[i] = f2bf(s[i]);
}

// ---------------------------------------------------------------------------
// GATv2 layer 1 (8 heads x 64 ch), one block per time step, 512 threads.
// Projections recomputed on the fly (D_IN=2). Output h1 as bf16 [t][n][c].
// ---------------------------------------------------------------------------
__global__ void gat1_kernel(const float* __restrict__ x,     // (T,33,2)
                            const int* __restrict__ ei,      // (2,97)
                            const float* __restrict__ ep1,   // (97,512)
                            const float* __restrict__ wl,    // (2,512)
                            const float* __restrict__ wr,    // (2,512)
                            const float* __restrict__ att,   // (8,64)
                            const float* __restrict__ bias,  // (512)
                            unsigned short* __restrict__ h1) // bf16 (T,33,512)
{
  extern __shared__ float sm[];
  float* s_wl    = sm;             // 1024
  float* s_wr    = s_wl + 1024;    // 1024
  float* s_x     = s_wr + 1024;    // 66
  float* s_logit = s_x + 66;       // 776
  float* s_m     = s_logit + 776;  // 264
  float* s_d     = s_m + 264;      // 264
  float* s_alpha = s_d + 264;      // 776
  float* s_out   = s_alpha + 776;  // 16896

  const int tid = threadIdx.x;
  const int t = blockIdx.x;

  s_wl[tid] = wl[tid]; s_wl[tid + 512] = wl[tid + 512];
  s_wr[tid] = wr[tid]; s_wr[tid + 512] = wr[tid + 512];
  if (tid < 66) s_x[tid] = x[t * 66 + tid];
  for (int i = tid; i < N_NODES * 512; i += 512) s_out[i] = 0.f;
  __syncthreads();

  // per-edge, per-head attention logits (one wave per edge, lanes over channels)
  const int wv = tid >> 5, ln = tid & 31;
  for (int e = wv; e < N_EDGES; e += 16) {
    int s = ei[e], d = ei[N_EDGES + e];
    float x0s = s_x[2 * s], x1s = s_x[2 * s + 1];
    float x0d = s_x[2 * d], x1d = s_x[2 * d + 1];
    for (int h = 0; h < 8; ++h) {
      float p = 0.f;
#pragma unroll
      for (int cc = 0; cc < 2; ++cc) {
        int hc = h * 64 + ln + cc * 32;
        float v = x0s * s_wl[hc] + x1s * s_wl[512 + hc]
                + x0d * s_wr[hc] + x1d * s_wr[512 + hc]
                + ep1[e * 512 + hc];
        v = v > 0.f ? v : 0.2f * v;  // leaky_relu(0.2)
        p += v * att[hc];
      }
#pragma unroll
      for (int off = 16; off > 0; off >>= 1) p += __shfl_down(p, off);
      if (ln == 0) s_logit[e * 8 + h] = p;
    }
  }
  __syncthreads();

  // segmented softmax over dst, per (node, head)
  if (tid < N_NODES * 8) {
    int n = tid >> 3, h = tid & 7;
    float m = -INFINITY;
    for (int e = 0; e < N_EDGES; ++e)
      if (ei[N_EDGES + e] == n) m = fmaxf(m, s_logit[e * 8 + h]);
    if (m == -INFINITY) m = 0.f;
    float dsum = 0.f;
    for (int e = 0; e < N_EDGES; ++e)
      if (ei[N_EDGES + e] == n) dsum += expf(s_logit[e * 8 + h] - m);
    s_m[tid] = m; s_d[tid] = dsum;
  }
  __syncthreads();

  for (int i = tid; i < N_EDGES * 8; i += 512) {
    int e = i >> 3, h = i & 7;
    int d = ei[N_EDGES + e];
    s_alpha[i] = expf(s_logit[i] - s_m[d * 8 + h]) / (s_d[d * 8 + h] + 1e-16f);
  }
  __syncthreads();

  // aggregate: each thread owns channel `tid` of every node (race-free)
  const int hc = tid, h = tid >> 6;
  for (int e = 0; e < N_EDGES; ++e) {
    int s = ei[e], d = ei[N_EDGES + e];
    float msg = s_x[2 * s] * s_wl[hc] + s_x[2 * s + 1] * s_wl[512 + hc];
    s_out[d * 512 + hc] += s_alpha[e * 8 + h] * msg;
  }
  float b = bias[hc];
  for (int n = 0; n < N_NODES; ++n) {
    float v = s_out[n * 512 + hc] + b;
    v = v > 0.f ? v : expf(v) - 1.f;  // ELU
    h1[((size_t)t * N_NODES + n) * 512 + hc] = f2bf(v);
  }
}

// ---------------------------------------------------------------------------
// bf16 WMMA GEMM: D(MxN,f32) = A(MxK,bf16 row-major) * Bt(NxK,bf16)^T
// One wave -> 16x16 tile. Block = 8 waves spanning 128 columns.
// Layouts per CDNA5 ISA 7.12.2 (wave32).
// ---------------------------------------------------------------------------
union V16 { v16bf v; uint4 q[2]; };

__global__ void gemm_bf16_kernel(const unsigned short* __restrict__ A,
                                 const unsigned short* __restrict__ Bt,
                                 float* __restrict__ D,
                                 int M, int N, int K)
{
  const int wave = threadIdx.x >> 5;
  const int lane = threadIdx.x & 31;
  const int l = lane & 15, half = lane >> 4;
  const int n0 = blockIdx.x * 128 + wave * 16;
  const int m0 = blockIdx.y * 16;
  if (n0 >= N || m0 >= M) return;  // wave-uniform; EXEC stays all-1s for WMMA

  v8f acc0 = {}, acc1 = {};
  const unsigned short* Arow = A + (size_t)(m0 + l) * K;
  const unsigned short* Brow = Bt + (size_t)(n0 + l) * K;

  for (int kc = 0; kc < K; kc += 64) {
    V16 a, b;
    // ---- chunk kc (even accumulator) ----
    int ka = kc + half * 8;          // A: lanes<16 K={kc..+7,kc+16..+23}; lanes>=16 +8
    a.q[0] = *(const uint4*)(Arow + ka);
    a.q[1] = *(const uint4*)(Arow + ka + 16);
    int kb = kc + half * 8;          // B col n0+l: lanes<16 K={kc..+7,kc+16..+23}; lanes>=16 +8
    b.q[0] = *(const uint4*)(Brow + kb);
    b.q[1] = *(const uint4*)(Brow + kb + 16);
    acc0 = __builtin_amdgcn_wmma_f32_16x16x32_bf16(false, a.v, false, b.v,
                                                   (short)0, acc0, false, false);
    // ---- chunk kc+32 (odd accumulator, hides WMMA RAW latency) ----
    ka += 32; kb += 32;
    a.q[0] = *(const uint4*)(Arow + ka);
    a.q[1] = *(const uint4*)(Arow + ka + 16);
    b.q[0] = *(const uint4*)(Brow + kb);
    b.q[1] = *(const uint4*)(Brow + kb + 16);
    acc1 = __builtin_amdgcn_wmma_f32_16x16x32_bf16(false, a.v, false, b.v,
                                                   (short)0, acc1, false, false);
  }

  float* Dp = D + (size_t)(m0 + half * 8) * N + n0 + l;
#pragma unroll
  for (int r = 0; r < 8; ++r) Dp[(size_t)r * N] = acc0[r] + acc1[r];
}

// ---------------------------------------------------------------------------
// GATv2 layer 2 attention (1 head x 512 ch). hl/hr precomputed by WMMA GEMM.
// Output written bf16 directly in [n][t][c] layout == reference's seq reshape.
// ---------------------------------------------------------------------------
__global__ void gat2_attn_kernel(const float* __restrict__ hl,   // (T,33,512)
                                 const float* __restrict__ hr,   // (T,33,512)
                                 const int* __restrict__ ei,
                                 const float* __restrict__ ep2,  // (97,512)
                                 const float* __restrict__ att2, // (512)
                                 const float* __restrict__ b2,   // (512)
                                 unsigned short* __restrict__ seq) // bf16 (33,T,512)
{
  extern __shared__ float sm[];
  float* s_logit = sm;            // 97
  float* s_m     = s_logit + 97;  // 33
  float* s_d     = s_m + 33;      // 33
  float* s_alpha = s_d + 33;      // 97
  float* s_out   = s_alpha + 97;  // 16896

  const int tid = threadIdx.x;
  const int t = blockIdx.x;
  for (int i = tid; i < N_NODES * 512; i += 512) s_out[i] = 0.f;

  const int wv = tid >> 5, ln = tid & 31;
  for (int e = wv; e < N_EDGES; e += 16) {
    int s = ei[e], d = ei[N_EDGES + e];
    const float* ps = hl + ((size_t)t * N_NODES + s) * 512;
    const float* pd = hr + ((size_t)t * N_NODES + d) * 512;
    const float* pe = ep2 + (size_t)e * 512;
    float p = 0.f;
#pragma unroll
    for (int cc = 0; cc < 16; ++cc) {
      int c = ln + cc * 32;
      float v = ps[c] + pd[c] + pe[c];
      v = v > 0.f ? v : 0.2f * v;
      p += v * att2[c];
    }
#pragma unroll
    for (int off = 16; off > 0; off >>= 1) p += __shfl_down(p, off);
    if (ln == 0) s_logit[e] = p;
  }
  __syncthreads();

  if (tid < N_NODES) {
    float m = -INFINITY;
    for (int e = 0; e < N_EDGES; ++e)
      if (ei[N_EDGES + e] == tid) m = fmaxf(m, s_logit[e]);
    if (m == -INFINITY) m = 0.f;
    float dsum = 0.f;
    for (int e = 0; e < N_EDGES; ++e)
      if (ei[N_EDGES + e] == tid) dsum += expf(s_logit[e] - m);
    s_m[tid] = m; s_d[tid] = dsum;
  }
  __syncthreads();
  if (tid < N_EDGES) {
    int d = ei[N_EDGES + tid];
    s_alpha[tid] = expf(s_logit[tid] - s_m[d]) / (s_d[d] + 1e-16f);
  }
  __syncthreads();

  for (int e = 0; e < N_EDGES; ++e) {
    int s = ei[e], d = ei[N_EDGES + e];
    s_out[d * 512 + tid] += s_alpha[e] * hl[((size_t)t * N_NODES + s) * 512 + tid];
  }
  float b = b2[tid];
  for (int n = 0; n < N_NODES; ++n) {
    float v = s_out[n * 512 + tid] + b;
    v = v > 0.f ? v : expf(v) - 1.f;  // ELU
    seq[((size_t)n * T_STEPS + t) * 512 + tid] = f2bf(v);
  }
}

// ---------------------------------------------------------------------------
// LSTM recurrences (input projections precomputed / folded). Single block.
// ---------------------------------------------------------------------------
__global__ void lstm1_kernel(const float* __restrict__ Z1,   // (T,512) = seq@wih^T
                             const float* __restrict__ whh,  // (512,128)
                             const float* __restrict__ bih,
                             const float* __restrict__ bhh,
                             float* __restrict__ hs)          // (T,128)
{
  __shared__ float s_h[HID1], s_c[HID1], s_z[512];
  const int tid = threadIdx.x;  // 512
  if (tid < HID1) { s_h[tid] = 0.f; s_c[tid] = 0.f; }
  const float bsum = bih[tid] + bhh[tid];
  const float* wr = whh + (size_t)tid * HID1;
  __syncthreads();
  for (int t = 0; t < T_STEPS; ++t) {
    float dot = 0.f;
#pragma unroll 8
    for (int j = 0; j < HID1; ++j) dot += s_h[j] * wr[j];
    s_z[tid] = Z1[t * 512 + tid] + dot + bsum;
    __syncthreads();
    if (tid < HID1) {
      float iv = s_z[tid], fv = s_z[HID1 + tid], gv = s_z[2 * HID1 + tid], ov = s_z[3 * HID1 + tid];
      float c = sigm(fv) * s_c[tid] + sigm(iv) * tanhf(gv);
      float h = sigm(ov) * tanhf(c);
      s_c[tid] = c; s_h[tid] = h;
      hs[t * HID1 + tid] = h;
    }
    __syncthreads();
  }
}

__global__ void lstm2_kernel(const float* __restrict__ hs1,  // (T,128)
                             const float* __restrict__ wih,  // (256,128)
                             const float* __restrict__ whh,  // (256,64)
                             const float* __restrict__ bih,
                             const float* __restrict__ bhh,
                             const float* __restrict__ fcw,  // (4,64)
                             const float* __restrict__ fcb,  // (4)
                             float* __restrict__ out)        // (4)
{
  __shared__ float s_h[HID2], s_c[HID2], s_z[4 * HID2], s_x[HID1];
  const int tid = threadIdx.x;  // 256
  if (tid < HID2) { s_h[tid] = 0.f; s_c[tid] = 0.f; }
  const float bsum = bih[tid] + bhh[tid];
  const float* wi = wih + (size_t)tid * HID1;
  const float* wh = whh + (size_t)tid * HID2;
  __syncthreads();
  for (int t = 0; t < T_STEPS; ++t) {
    if (tid < HID1) s_x[tid] = hs1[t * HID1 + tid];
    __syncthreads();
    float dot = 0.f;
#pragma unroll 8
    for (int j = 0; j < HID1; ++j) dot += s_x[j] * wi[j];
#pragma unroll 8
    for (int j = 0; j < HID2; ++j) dot += s_h[j] * wh[j];
    s_z[tid] = dot + bsum;
    __syncthreads();
    if (tid < HID2) {
      float iv = s_z[tid], fv = s_z[HID2 + tid], gv = s_z[2 * HID2 + tid], ov = s_z[3 * HID2 + tid];
      float c = sigm(fv) * s_c[tid] + sigm(iv) * tanhf(gv);
      float h = sigm(ov) * tanhf(c);
      s_c[tid] = c; s_h[tid] = h;
    }
    __syncthreads();
  }
  if (tid < 4) {
    float o = fcb[tid];
    for (int j = 0; j < HID2; ++j) o += s_h[j] * fcw[tid * HID2 + j];
    out[tid] = o;
  }
}

// ---------------------------------------------------------------------------
extern "C" void kernel_launch(void* const* d_in, const int* in_sizes, int n_in,
                              void* d_out, int out_size, void* d_ws, size_t ws_size,
                              hipStream_t stream)
{
  (void)in_sizes; (void)n_in; (void)out_size; (void)ws_size;
  const float* x      = (const float*)d_in[0];
  const int*   ei     = (const int*)  d_in[1];
  const float* ea     = (const float*)d_in[2];
  const float* g1_wl  = (const float*)d_in[3];
  const float* g1_wr  = (const float*)d_in[4];
  const float* g1_we  = (const float*)d_in[5];
  const float* g1_att = (const float*)d_in[6];
  const float* g1_b   = (const float*)d_in[7];
  const float* g2_wl  = (const float*)d_in[8];
  const float* g2_wr  = (const float*)d_in[9];
  const float* g2_we  = (const float*)d_in[10];
  const float* g2_att = (const float*)d_in[11];
  const float* g2_b   = (const float*)d_in[12];
  const float* l1_wih = (const float*)d_in[13];
  const float* l1_whh = (const float*)d_in[14];
  const float* l1_bih = (const float*)d_in[15];
  const float* l1_bhh = (const float*)d_in[16];
  const float* l2_wih = (const float*)d_in[17];
  const float* l2_whh = (const float*)d_in[18];
  const float* l2_bih = (const float*)d_in[19];
  const float* l2_bhh = (const float*)d_in[20];
  const float* fc_w   = (const float*)d_in[21];
  const float* fc_b   = (const float*)d_in[22];

  char* base = (char*)d_ws; size_t off = 0;
  auto carve = [&](size_t bytes) -> void* {
    void* p = base + off; off += (bytes + 255) & ~(size_t)255; return p;
  };
  const size_t TN = (size_t)T_STEPS * N_NODES;                       // 33792
  unsigned short* h1   = (unsigned short*)carve(TN * 512 * 2);       // bf16 GAT1 out
  unsigned short* wlT  = (unsigned short*)carve((size_t)512 * 512 * 2);
  unsigned short* wrT  = (unsigned short*)carve((size_t)512 * 512 * 2);
  unsigned short* wih1 = (unsigned short*)carve((size_t)512 * 16896 * 2);
  unsigned short* seqb = (unsigned short*)carve(TN * 512 * 2);       // bf16 [n][t][c]
  float* hl  = (float*)carve(TN * 512 * 4);
  float* hr  = (float*)carve(TN * 512 * 4);
  float* ep1 = (float*)carve((size_t)N_EDGES * 512 * 4);
  float* ep2 = (float*)carve((size_t)N_EDGES * 512 * 4);
  float* Z1  = (float*)carve((size_t)T_STEPS * 512 * 4);
  float* hs1 = (float*)carve((size_t)T_STEPS * HID1 * 4);

  // Prep: edge projections + weight conversions (bf16, B pre-transposed)
  edge_proj_kernel<<<N_EDGES, 512, 0, stream>>>(ea, g1_we, ep1);
  edge_proj_kernel<<<N_EDGES, 512, 0, stream>>>(ea, g2_we, ep2);
  convert_transpose_512<<<512, 512, 0, stream>>>(g2_wl, wlT);
  convert_transpose_512<<<512, 512, 0, stream>>>(g2_wr, wrT);
  convert_bf16_kernel<<<(512 * 16896 + 255) / 256, 256, 0, stream>>>(l1_wih, wih1, 512 * 16896);

  // GAT layer 1 (fused per-t)
  gat1_kernel<<<T_STEPS, 512, 21090 * 4, stream>>>(x, ei, ep1, g1_wl, g1_wr, g1_att, g1_b, h1);

  // GAT layer 2 projections: (33792,512)@(512,512) x2 via bf16 WMMA
  dim3 gA(512 / 128, (unsigned)(TN / 16));
  gemm_bf16_kernel<<<gA, 256, 0, stream>>>(h1, wlT, hl, (int)TN, 512, 512);
  gemm_bf16_kernel<<<gA, 256, 0, stream>>>(h1, wrT, hr, (int)TN, 512, 512);

  // GAT layer 2 attention + ELU, writes seq in [n][t][c] (== reference reshape)
  gat2_attn_kernel<<<T_STEPS, 512, 17156 * 4, stream>>>(hl, hr, ei, ep2, g2_att, g2_b, seqb);

  // LSTM1 input projection: (1024,16896)@(16896,512) via bf16 WMMA
  dim3 gB(512 / 128, T_STEPS / 16);
  gemm_bf16_kernel<<<gB, 256, 0, stream>>>(seqb, wih1, Z1, T_STEPS, 512, 16896);

  // Sequential recurrences + FC head
  lstm1_kernel<<<1, 512, 0, stream>>>(Z1, l1_whh, l1_bih, l1_bhh, hs1);
  lstm2_kernel<<<1, 256, 0, stream>>>(hs1, l2_wih, l2_whh, l2_bih, l2_bhh, fc_w, fc_b, (float*)d_out);
}
